// OffsetsPredictor_3109556323110
// MI455X (gfx1250) — matrix operand order repaired and checked by reading, not compiled, not run on hardware
//
#include <hip/hip_runtime.h>

// ---------------------------------------------------------------------------
// OffsetsPredictor for MI455X (gfx1250): bf16 WMMA implicit-GEMM ConvLSTM.
//   1. stats_kernel    : 5 scalar moments of pre_offset (BN collapses to affine)
//   2. coeff_kernel    : per-channel (alpha,beta,delta) folding conv+BN
//   3. zero_border     : zero the 1-px halo ring of the packed activations
//   4. pack_combined   : channel-last bf16 [B][258][258][48] = cat(x, emb, h0)
//   5. pack_weights    : lstm_W -> bf16 [64][448] (K = tap*48+ch, padded)
//   6. conv_lstm_wmma  : 3x3 conv 48->64 via v_wmma_f32_16x16x32_bf16;
//                        A: double-buffered global_load_b128,
//                        B: LDS-resident via asm ds_load_b128 (keeps VGPRs low
//                        -> high occupancy instead of 448 reg-resident frags),
//                        fused gates + cell update + 1x1 offset head
//   7. bilinear_kernel : clamped bilinear gather of x at grid+offset
// ---------------------------------------------------------------------------

#define BATCH 16
#define HH 256
#define WW 256
#define HW (HH * WW)
#define HP 258           // halo-padded spatial dim
#define KVALID 432       // 9 taps * 48 channels
#define KPAD 448         // 14 * 32
#define KSTEPS 14
#define LDSTRIDE 228     // dword row stride in LDS: 16B aligned, bank-spread

typedef __attribute__((ext_vector_type(16))) __bf16 v16bf;
typedef __attribute__((ext_vector_type(8)))  float  v8f;
typedef __attribute__((ext_vector_type(4)))  unsigned int v4u;

union Frag { unsigned int u[8]; v16bf v; };
union Pix  { unsigned int u[24]; uint4 q[6]; };

__device__ __forceinline__ unsigned short f2bf(float f) {
    unsigned int u = __float_as_uint(f);
    unsigned int r = u + 0x7FFFu + ((u >> 16) & 1u);   // round-to-nearest-even
    return (unsigned short)(r >> 16);
}
__device__ __forceinline__ float fsigmoid(float x) {
    return 1.f / (1.f + __expf(-x));
}
__device__ __forceinline__ float ftanh(float x) {
    float e = __expf(-2.f * fabsf(x));                 // in (0,1], NaN-safe
    float r = (1.f - e) / (1.f + e);
    return copysignf(r, x);
}

// ---------------- 1. moments of pre_offset ---------------------------------
__global__ __launch_bounds__(256) void stats_kernel(
    const float* __restrict__ pre, float* __restrict__ stats)
{
    __shared__ float sh[5 * 256];
    float s0 = 0.f, s1 = 0.f, s00 = 0.f, s11 = 0.f, s01 = 0.f;
    const int N = BATCH * HW;
    for (int i = blockIdx.x * blockDim.x + threadIdx.x; i < N;
         i += gridDim.x * blockDim.x) {
        int b = i >> 16, rem = i & 65535;
        float a = pre[(b * 2 + 0) * HW + rem];
        float c = pre[(b * 2 + 1) * HW + rem];
        s0 += a; s1 += c; s00 += a * a; s11 += c * c; s01 += a * c;
    }
    int t = threadIdx.x;
    sh[t] = s0; sh[256 + t] = s1; sh[512 + t] = s00;
    sh[768 + t] = s11; sh[1024 + t] = s01;
    __syncthreads();
    for (int ofs = 128; ofs > 0; ofs >>= 1) {
        if (t < ofs)
            for (int q = 0; q < 5; ++q)
                sh[q * 256 + t] += sh[q * 256 + t + ofs];
        __syncthreads();
    }
    if (t == 0)
        for (int q = 0; q < 5; ++q)
            atomicAdd(&stats[q], sh[q * 256]);
}

// ---------------- 2. fold 1x1 conv + BN + bias into per-channel affine -----
__global__ void coeff_kernel(
    const float* __restrict__ stats, const float* __restrict__ embW,
    const float* __restrict__ embB, const float* __restrict__ gamma,
    const float* __restrict__ bnbeta, float* __restrict__ coef)
{
    int c = threadIdx.x;
    if (c >= 16) return;
    const float N = (float)(BATCH * HW);
    float mu0 = stats[0] / N, mu1 = stats[1] / N;
    float v0 = stats[2] / N - mu0 * mu0;
    float v1 = stats[3] / N - mu1 * mu1;
    float cv = stats[4] / N - mu0 * mu1;
    float a = embW[c * 2 + 0], b = embW[c * 2 + 1];
    float mean = a * mu0 + b * mu1 + embB[c];
    float var  = a * a * v0 + b * b * v1 + 2.f * a * b * cv;
    float sc = gamma[c] * rsqrtf(var + 1e-5f);
    coef[c]      = sc * a;                              // alpha
    coef[16 + c] = sc * b;                              // beta
    coef[32 + c] = sc * (embB[c] - mean) + bnbeta[c];   // delta
}

// ---------------- 3. zero the halo ring ------------------------------------
__global__ __launch_bounds__(256) void zero_border(uint4* __restrict__ comb4)
{
    int i = blockIdx.x * blockDim.x + threadIdx.x;   // BATCH * 1028 border px
    int b = i / 1028;
    if (b >= BATCH) return;
    int j = i - b * 1028;
    int hy, hx;
    if (j < 258)      { hy = 0;        hx = j; }
    else if (j < 516) { hy = 257;      hx = j - 258; }
    else if (j < 772) { hy = j - 515;  hx = 0; }       // 1..256
    else              { hy = j - 771;  hx = 257; }     // 1..256
    uint4* dst = comb4 + ((size_t)(b * HP + hy) * HP + hx) * 6;
    uint4 z = {0u, 0u, 0u, 0u};
#pragma unroll
    for (int q = 0; q < 6; ++q) dst[q] = z;
}

// ---------------- 4. channel-last bf16 activations (halo interior) ---------
__global__ __launch_bounds__(256) void pack_combined(
    const float* __restrict__ x, const float* __restrict__ pre,
    const float* __restrict__ h0, const float* __restrict__ coef,
    uint4* __restrict__ comb4)   // [B][258][258][6] uint4 (48 bf16/pixel)
{
    int p = blockIdx.x * blockDim.x + threadIdx.x;
    if (p >= BATCH * HW) return;
    int b = p >> 16, rem = p & 65535;
    int y = rem >> 8, xx = rem & 255;
    float p0 = pre[(b * 2 + 0) * HW + rem];
    float p1 = pre[(b * 2 + 1) * HW + rem];
    Pix out;
#pragma unroll
    for (int j = 0; j < 8; ++j) {
        float a0 = x[(b * 16 + 2 * j)     * HW + rem];
        float a1 = x[(b * 16 + 2 * j + 1) * HW + rem];
        out.u[j] = (unsigned)f2bf(a0) | ((unsigned)f2bf(a1) << 16);
    }
#pragma unroll
    for (int j = 0; j < 8; ++j) {
        int ca = 2 * j, cb = 2 * j + 1;
        float e0 = coef[ca] * p0 + coef[16 + ca] * p1 + coef[32 + ca];
        float e1 = coef[cb] * p0 + coef[16 + cb] * p1 + coef[32 + cb];
        e0 = e0 > 0.f ? e0 : 0.2f * e0;                 // LeakyReLU(0.2)
        e1 = e1 > 0.f ? e1 : 0.2f * e1;
        out.u[8 + j] = (unsigned)f2bf(e0) | ((unsigned)f2bf(e1) << 16);
    }
#pragma unroll
    for (int j = 0; j < 8; ++j) {
        float a0 = h0[(b * 16 + 2 * j)     * HW + rem];
        float a1 = h0[(b * 16 + 2 * j + 1) * HW + rem];
        out.u[16 + j] = (unsigned)f2bf(a0) | ((unsigned)f2bf(a1) << 16);
    }
    uint4* dst = comb4 + ((size_t)(b * HP + y + 1) * HP + (xx + 1)) * 6;
#pragma unroll
    for (int j = 0; j < 6; ++j) dst[j] = out.q[j];
}

// ---------------- 5. bf16 weights, K = tap*48 + ch, padded to 448 ----------
__global__ __launch_bounds__(256) void pack_weights(
    const float* __restrict__ lstmW, unsigned short* __restrict__ wp)
{
    int i = blockIdx.x * blockDim.x + threadIdx.x;
    if (i >= 64 * KPAD) return;
    int n = i / KPAD, k = i - n * KPAD;
    unsigned short v = 0;
    if (k < KVALID) {
        int t = k / 48, c = k - t * 48;
        v = f2bf(lstmW[(n * 48 + c) * 9 + t]);          // OIHW, t = ty*3+tx
    }
    wp[i] = v;
}

// ---------------- A-fragment loader (constant-folds after unroll) ----------
__device__ __forceinline__ void load_a(Frag& a, const unsigned int* __restrict__ comb,
                                       const unsigned (&tb)[9], int hf, int ks)
{
#pragma unroll
    for (int g = 0; g < 2; ++g) {
        const int k0 = ks * 32 + g * 16;                // lane-k for hf=0
        if (k0 <= 416) {                                // uniform pad validity
            const int k1 = k0 + 8;                      // lane-k for hf=1
            const int t0 = k0 / 48, d0 = (k0 % 48) >> 1;
            const int t1 = k1 / 48, d1 = (k1 % 48) >> 1;
            const unsigned a0 = tb[t0] + (unsigned)d0;
            const unsigned a1 = tb[t1] + (unsigned)d1;
            const uint4 v = *(const uint4*)(comb + (hf ? a1 : a0));
            a.u[g * 4 + 0] = v.x; a.u[g * 4 + 1] = v.y;
            a.u[g * 4 + 2] = v.z; a.u[g * 4 + 3] = v.w;
        } else {                                        // only (ks=13, g=1)
            a.u[g * 4 + 0] = 0u; a.u[g * 4 + 1] = 0u;
            a.u[g * 4 + 2] = 0u; a.u[g * 4 + 3] = 0u;
        }
    }
}

// ---------------- 6. WMMA conv + gates + cell + 1x1 head -------------------
__global__ __launch_bounds__(256) void conv_lstm_wmma(
    const unsigned int* __restrict__ comb,       // halo [B][258][258][24] dw
    const unsigned int* __restrict__ wpack,      // [64][224] dw (bf16x2)
    const float* __restrict__ c0,
    const float* __restrict__ outW,              // (2,16)
    const float* __restrict__ outB,              // (2,)
    float* __restrict__ off)                     // (B,2,H,W)
{
    __shared__ unsigned int Wlds[64 * LDSTRIDE]; // 58368 B, padded stride
    for (int i = threadIdx.x; i < 64 * 224; i += 256) {
        int n = i / 224, j = i - n * 224;
        Wlds[n * LDSTRIDE + j] = wpack[i];
    }
    __syncthreads();

    const int lane = threadIdx.x & 31;
    const int hf = lane >> 4;                    // half-wave id
    const int ln = lane & 15;

    // one wave = one strip: 8 tiles (128 consecutive pixels) of one row
    const int strip = blockIdx.x * 8 + (threadIdx.x >> 5);  // 8192 strips
    const int sx0 = (strip & 1) << 7;            // 0 or 128
    const int y   = (strip >> 1) & 255;
    const int b   = strip >> 9;

    const float w0 = outW[ln];                   // out_W[0][ch]
    const float w1 = outW[16 + ln];              // out_W[1][ch]
    const float ob0 = outB[0], ob1 = outB[1];

    // per-lane LDS byte addresses for the 4 B-fragment columns (hoistable)
    unsigned wb[4];
#pragma unroll
    for (int nt = 0; nt < 4; ++nt)
        wb[nt] = (unsigned)(size_t)&Wlds[(nt * 16 + ln) * LDSTRIDE + (hf << 2)];

    for (int tt = 0; tt < 8; ++tt) {
        const int tx0 = sx0 + (tt << 4);
        const int xA = tx0 + ln;                 // A-matrix row (pixel M = ln)

        // 9 tap pixel base dword addresses in the halo buffer; always
        // in-bounds, so zero predication in the K loop.
        unsigned tb[9];
#pragma unroll
        for (int t = 0; t < 9; ++t) {
            int hy = y + (t / 3);                // (y-1+ty) + 1 halo shift
            int hx = xA + (t % 3);               // (xA-1+tx) + 1 halo shift
            tb[t] = ((unsigned)(b * HP + hy) * HP + (unsigned)hx) * 24u;
        }

        v8f acc[4] = {};                         // N-tiles = gates i,f,o,g
        Frag aCur, aNxt;
        load_a(aCur, comb, tb, hf, 0);
#pragma unroll
        for (int ks = 0; ks < KSTEPS; ++ks) {
            if (ks + 1 < KSTEPS)                 // prefetch next A fragment
                load_a(aNxt, comb, tb, hf, ks + 1);
            // B fragments: volatile ds_load_b128 pairs -> stay LDS-resident
            // (prevents the compiler hoisting 448 VGPRs of weights, keeps
            // occupancy at ~8 waves/SIMD instead of 1-2)
#pragma unroll
            for (int nt = 0; nt < 4; ++nt) {
                v4u lo, hi;
                asm volatile(
                    "ds_load_b128 %0, %2 offset:%3\n\t"
                    "ds_load_b128 %1, %2 offset:%4\n\t"
                    "s_wait_dscnt 0x0"
                    : "=v"(lo), "=v"(hi)
                    : "v"(wb[nt]), "i"(ks * 64), "i"(ks * 64 + 32));
                Frag bm;
                bm.u[0] = lo[0]; bm.u[1] = lo[1]; bm.u[2] = lo[2]; bm.u[3] = lo[3];
                bm.u[4] = hi[0]; bm.u[5] = hi[1]; bm.u[6] = hi[2]; bm.u[7] = hi[3];
                acc[nt] = __builtin_amdgcn_wmma_f32_16x16x32_bf16(
                    false, aCur.v, false, bm.v, (short)0, acc[nt], false, false);
            }
            if (ks + 1 < KSTEPS) aCur = aNxt;
        }

        // ---- epilogue: gates, cell update, 1x1 offset head --------------
#pragma unroll
        for (int r = 0; r < 8; ++r) {
            int px = tx0 + r + (hf << 3);        // C-matrix: pixel = r + 8*hf
            float ig = fsigmoid(acc[0][r]);
            float fg = fsigmoid(acc[1][r]);
            float og = fsigmoid(acc[2][r]);
            float gg = ftanh(acc[3][r]);
            float c0v = c0[((size_t)(b * 16 + ln) * HH + y) * WW + px];
            float c1 = fg * c0v + ig * gg;
            float h1 = og * ftanh(c1);
            float s0 = h1 * w0;                  // channel = ln; 16-lane sum
            float s1 = h1 * w1;
#pragma unroll
            for (int d = 1; d < 16; d <<= 1) {   // stays within each half
                s0 += __shfl_xor(s0, d, 32);
                s1 += __shfl_xor(s1, d, 32);
            }
            if (ln == 0) {
                off[((size_t)(b * 2 + 0) * HH + y) * WW + px] = s0 + ob0;
                off[((size_t)(b * 2 + 1) * HH + y) * WW + px] = s1 + ob1;
            }
        }
    }
}

// ---------------- 7. clamped bilinear gather -------------------------------
__global__ __launch_bounds__(256) void bilinear_kernel(
    const float* __restrict__ x, const float* __restrict__ off,
    float* __restrict__ out)
{
    int i = blockIdx.x * blockDim.x + threadIdx.x;  // B*C*H*W = 2^24
    if (i >= BATCH * 16 * HW) return;
    int pos = i & 65535;
    int bc = i >> 16;
    int b = bc >> 4;
    int yi = pos >> 8, xi = pos & 255;
    float dy = off[((size_t)(b * 2 + 0) << 16) + pos];
    float dx = off[((size_t)(b * 2 + 1) << 16) + pos];
    float yc = fminf(fmaxf((float)yi + dy, 0.f), 255.f);
    float xc = fminf(fmaxf((float)xi + dx, 0.f), 255.f);
    float y0f = floorf(yc), x0f = floorf(xc);
    int y0 = (int)y0f,       x0 = (int)x0f;
    int y1 = (int)ceilf(yc), x1 = (int)ceilf(xc);   // ceil, as reference
    const float* img = x + ((size_t)bc << 16);
    float v00 = img[y0 * WW + x0], v01 = img[y0 * WW + x1];
    float v10 = img[y1 * WW + x0], v11 = img[y1 * WW + x1];
    float fdx = xc - x0f, fdy = yc - y0f;
    float top = v00 + (v01 - v00) * fdx;
    float bot = v10 + (v11 - v10) * fdx;
    out[i] = top + (bot - top) * fdy;
}

// ---------------------------------------------------------------------------
extern "C" void kernel_launch(void* const* d_in, const int* in_sizes, int n_in,
                              void* d_out, int out_size, void* d_ws, size_t ws_size,
                              hipStream_t stream) {
    const float* x     = (const float*)d_in[0];
    const float* pre   = (const float*)d_in[1];
    const float* h0    = (const float*)d_in[2];
    const float* c0    = (const float*)d_in[3];
    const float* embW  = (const float*)d_in[4];
    const float* embB  = (const float*)d_in[5];
    const float* gamma = (const float*)d_in[6];
    const float* bnb   = (const float*)d_in[7];
    const float* lstmW = (const float*)d_in[8];
    const float* outW  = (const float*)d_in[9];
    const float* outB  = (const float*)d_in[10];
    float* out = (float*)d_out;

    // workspace layout (bytes)
    char* ws = (char*)d_ws;
    float*          stats = (float*)ws;                       // 5 f32 (pad 64B)
    float*          coef  = (float*)(ws + 64);                // 48 f32
    unsigned short* wp    = (unsigned short*)(ws + 256);      // 57,344 B
    unsigned int*   comb  = (unsigned int*)(ws + 57600);      // 102,242,304 B
    float*          offb  = (float*)(ws + 57600 + 102242304); // 8,388,608 B

    hipMemsetAsync(stats, 0, 64, stream);
    stats_kernel   <<<1024, 256, 0, stream>>>(pre, stats);
    coeff_kernel   <<<1, 16, 0, stream>>>(stats, embW, embB, gamma, bnb, coef);
    zero_border    <<<(BATCH * 1028 + 255) / 256, 256, 0, stream>>>((uint4*)comb);
    pack_combined  <<<(BATCH * HW) / 256, 256, 0, stream>>>(x, pre, h0, coef,
                                                            (uint4*)comb);
    pack_weights   <<<(64 * KPAD) / 256, 256, 0, stream>>>(lstmW, wp);
    conv_lstm_wmma <<<1024, 256, 0, stream>>>(comb, (const unsigned int*)wp,
                                              c0, outW, outB, offb);
    bilinear_kernel<<<(BATCH * 16 * HW) / 256, 256, 0, stream>>>(x, offb, out);
}
